// PointMVSNet_41257455845937
// MI455X (gfx1250) — matrix operand order repaired and checked by reading, not compiled, not run on hardware
//
#include <hip/hip_runtime.h>
#include <hip/hip_bf16.h>

// ---------------------------------------------------------------------------
// PointMVSNet forward for MI455X (gfx1250, wave32, WMMA + TDM).
// Convs run as implicit-GEMM through v_wmma_f32_16x16x32_f16 (f32 accumulate).
// Weight tiles are staged into LDS by the Tensor Data Mover (tensor_load_to_lds
// with LDS padding), activations are gathered im2col-style with a division-free
// incremental decode. Geometry & softmax stay in f32.
// Block = 256 threads (8 waves) per the CDNA5 WGP guidance.
// ---------------------------------------------------------------------------

typedef __attribute__((ext_vector_type(16))) _Float16 v16h;
typedef __attribute__((ext_vector_type(8)))  float    v8f;
typedef __attribute__((ext_vector_type(4)))  unsigned int u32x4;
typedef __attribute__((ext_vector_type(8)))  int          i32x8;
typedef __attribute__((ext_vector_type(4)))  int          i32x4;

#if defined(__has_builtin)
#if __has_builtin(__builtin_amdgcn_tensor_load_to_lds)
#define HAVE_TDM 1
#pragma message("gfx1250: TDM tensor_load_to_lds path ENABLED")
#endif
#endif
#ifndef HAVE_TDM
#define HAVE_TDM 0
#pragma message("TDM builtin not available on this pass; cooperative-load fallback")
#endif

// ----------------------------- small kernels -------------------------------

__global__ void k_f32_to_f16(const float* __restrict__ s, _Float16* __restrict__ d, int n) {
  int i = blockIdx.x * blockDim.x + threadIdx.x;
  if (i < n) d[i] = (_Float16)s[i];
}

__global__ void k_add_f16(const _Float16* __restrict__ a, const _Float16* __restrict__ b,
                          _Float16* __restrict__ o, int n) {
  int i = blockIdx.x * blockDim.x + threadIdx.x;
  if (i < n) o[i] = (_Float16)((float)a[i] + (float)b[i]);
}

// ----------------------------- geometry setup ------------------------------

__device__ static void inv3(const float a[3][3], float o[3][3]) {
  float c00 =  (a[1][1]*a[2][2] - a[1][2]*a[2][1]);
  float c01 = -(a[1][0]*a[2][2] - a[1][2]*a[2][0]);
  float c02 =  (a[1][0]*a[2][1] - a[1][1]*a[2][0]);
  float det = a[0][0]*c00 + a[0][1]*c01 + a[0][2]*c02;
  float id = 1.0f / det;
  o[0][0] =  (a[1][1]*a[2][2] - a[1][2]*a[2][1]) * id;
  o[0][1] = -(a[0][1]*a[2][2] - a[0][2]*a[2][1]) * id;
  o[0][2] =  (a[0][1]*a[1][2] - a[0][2]*a[1][1]) * id;
  o[1][0] = -(a[1][0]*a[2][2] - a[1][2]*a[2][0]) * id;
  o[1][1] =  (a[0][0]*a[2][2] - a[0][2]*a[2][0]) * id;
  o[1][2] = -(a[0][0]*a[1][2] - a[0][2]*a[1][0]) * id;
  o[2][0] =  (a[1][0]*a[2][1] - a[1][1]*a[2][0]) * id;
  o[2][1] = -(a[0][0]*a[2][1] - a[0][1]*a[2][0]) * id;
  o[2][2] =  (a[0][0]*a[1][1] - a[0][1]*a[1][0]) * id;
}
__device__ static void mm3(const float a[3][3], const float b[3][3], float o[3][3]) {
  for (int i = 0; i < 3; ++i)
    for (int j = 0; j < 3; ++j)
      o[i][j] = a[i][0]*b[0][j] + a[i][1]*b[1][j] + a[i][2]*b[2][j];
}
__device__ static void mv3(const float a[3][3], const float v[3], float o[3]) {
  for (int i = 0; i < 3; ++i) o[i] = a[i][0]*v[0] + a[i][1]*v[1] + a[i][2]*v[2];
}

// cams layout: (B=1, V=3, 2, 4, 4). Produces per-view affine uvz = d*A_v*p + b_v.
// pb: [v*12+0..8]=A_v, [v*12+9..11]=b_v, pb[36]=depth_start, pb[37]=depth_interval.
__global__ void k_setup(const float* __restrict__ cams, float* __restrict__ pb) {
  if (threadIdx.x != 0 || blockIdx.x != 0) return;
  float R[3][3][3], t[3][3], Km[3][3][3];
  for (int v = 0; v < 3; ++v) {
    for (int r = 0; r < 3; ++r) {
      for (int c = 0; c < 3; ++c) {
        R[v][r][c] = cams[((v*2 + 0)*4 + r)*4 + c];
        float kv   = cams[((v*2 + 1)*4 + r)*4 + c];
        if (r < 2) kv *= 0.5f;                       // K[:, :, :2, :3] /= 2
        Km[v][r][c] = kv;
      }
      t[v][r] = cams[((v*2 + 0)*4 + r)*4 + 3];
    }
  }
  float K0i[3][3], R0i[3][3], G0[3][3], w0[3];
  inv3(Km[0], K0i);
  inv3(R[0],  R0i);
  mm3(R0i, K0i, G0);       // world = d*G0*p - R0i*t0
  mv3(R0i, t[0], w0);
  for (int v = 0; v < 3; ++v) {
    float M[3][3], A[3][3], Mw[3], Kt[3];
    mm3(Km[v], R[v], M);
    mm3(M, G0, A);
    mv3(M, w0, Mw);
    mv3(Km[v], t[v], Kt);
    for (int i = 0; i < 9; ++i) pb[v*12 + i] = A[i/3][i%3];
    for (int i = 0; i < 3; ++i) pb[v*12 + 9 + i] = Kt[i] - Mw[i];
  }
  pb[36] = cams[(0*2 + 1)*16 + 3*4 + 0];   // depth_start
  pb[37] = cams[(0*2 + 1)*16 + 3*4 + 1];   // depth_interval
}

// ----------------------- WMMA implicit-GEMM convolution --------------------
// GEMM: M = Co (tile 16), N = OD*OH*OW (tile 256/block: 8 waves x 2 subtiles),
// K = Ci*KD*KH*KW stepped by 64 (two WMMA k-chunks per step).

struct ConvDims {
  int Ci, Co;
  int KD, KH, KW;
  int ID, IH, IW;
  int OD, OH, OW;
  int SD, SH, SW;
  int PD, PH, PW;
  int relu;
};

__global__ __launch_bounds__(256) void conv_wmma_kernel(
    const _Float16* __restrict__ in,      // [Ci][ID][IH][IW]
    const _Float16* __restrict__ wgt,     // [Co][K]
    const float*    __restrict__ bias,    // [Co]
    _Float16*       __restrict__ out16,   // [Co][OD][OH][OW] (optional)
    float*          __restrict__ out32,   // (optional)
    ConvDims cd)
{
  const int K     = cd.Ci * cd.KD * cd.KH * cd.KW;
  const int NSP   = cd.OD * cd.OH * cd.OW;
  const int tileN = blockIdx.x * 256;
  const int m0    = blockIdx.y * 16;
  const int tid   = threadIdx.x;
  const int lane  = tid & 31;
  const int wave  = tid >> 5;

  // Row stride 68 halves (136B) = 128B data + 8B pad; matches TDM pad config.
  __shared__ _Float16 aT[16][68];    // weights tile  [m][k]     (16 x 64)
  __shared__ _Float16 bT[256][68];   // im2col tile   [n_loc][k] (256 x 64)

  v8f acc0 = {};
  v8f acc1 = {};

  // Per-thread B-tile pixel (one pixel per thread, all 64 k-values).
  const int  opix   = tileN + tid;
  const bool pvalid = opix < NSP;
  int oz = 0, oy = 0, ox = 0;
  if (pvalid) {
    oz = opix / (cd.OH * cd.OW);
    int r = opix - oz * cd.OH * cd.OW;
    oy = r / cd.OW;
    ox = r - oy * cd.OW;
  }
  const int izb = oz * cd.SD - cd.PD;
  const int iyb = oy * cd.SH - cd.PH;
  const int ixb = ox * cd.SW - cd.PW;

  // Division-free im2col decode state, carried across all K-steps.
  int kx = 0, ky = 0, kd = 0, ci = 0;

  for (int k0 = 0; k0 < K; k0 += 64) {
    // ---------------- stage A (16 x 64 halves) ----------------
#if HAVE_TDM
    if (wave == 0) {
      // TDM descriptor (CDNA5 ISA):
      // 2D tile 64 x 16, element 2B, row stride K elements, zero-fill OOB,
      // LDS padding: 2 DWORDs after every 32 DWORDs -> 136B LDS row stride.
      unsigned long long ga =
          (unsigned long long)(const void*)(wgt + (unsigned long long)m0 * K + k0);
      unsigned ldsa = (unsigned)(unsigned long long)(const void*)&aT[0][0];
      int tdim0 = K - k0;            // remaining columns (reads past -> 0)
      int tdim1 = cd.Co - m0;        // remaining rows    (reads past -> 0)
      u32x4 g0;
      g0[0] = 1u;                                          // count=1, user mode
      g0[1] = ldsa;                                        // lds_addr
      g0[2] = (unsigned)(ga & 0xffffffffull);              // global_addr lo
      g0[3] = (unsigned)((ga >> 32) & 0x01ffffffull) | (2u << 30);  // hi | type=2
      i32x8 g1;
      g1[0] = (1 << 16)        // data_size = 2B
            | (1 << 20)        // pad_enable
            | (4 << 22)        // pad_interval: 32 DWORDs (128B)
            | (1 << 25);       // pad_amount:   2 DWORDs (8B)
      g1[1] = (tdim0 & 0xffff) << 16;                       // tensor_dim0[15:0]
      g1[2] = ((tdim0 >> 16) & 0xffff) | ((tdim1 & 0xffff) << 16);
      g1[3] = ((tdim1 >> 16) & 0xffff) | (64 << 16);        // tile_dim0 = 64
      g1[4] = 16;                                           // tile_dim1 = 16
      g1[5] = K;                                            // dim0_stride lo32
      g1[6] = 0;
      g1[7] = 0;
      i32x4 gz = {0, 0, 0, 0};
#if __clang_major__ >= 23
      i32x8 gz8 = {0, 0, 0, 0, 0, 0, 0, 0};
      __builtin_amdgcn_tensor_load_to_lds(g0, g1, gz, gz, gz8, 0);
#else
      __builtin_amdgcn_tensor_load_to_lds(g0, g1, gz, gz, 0);
#endif
      __builtin_amdgcn_s_wait_tensorcnt(0);
    }
#else
    {
      int q  = tid * 4;          // 256 threads x 4 halves = 16x64
      int m  = q >> 6;
      int kk = q & 63;
      int gm = m0 + m;
      #pragma unroll
      for (int j = 0; j < 4; ++j) {
        int gk = k0 + kk + j;
        _Float16 v = (_Float16)0.f;
        if (gm < cd.Co && gk < K) v = wgt[(unsigned long long)gm * K + gk];
        aT[m][kk + j] = v;
      }
    }
#endif

    // ------------- stage B (256 x 64 halves), incremental decode -----------
    for (int j = 0; j < 64; ++j) {
      _Float16 v = (_Float16)0.f;
      if (pvalid && (k0 + j) < K) {
        int iz = izb + kd, iy = iyb + ky, ix = ixb + kx;
        if ((unsigned)iz < (unsigned)cd.ID && (unsigned)iy < (unsigned)cd.IH &&
            (unsigned)ix < (unsigned)cd.IW)
          v = in[((unsigned long long)(ci * cd.ID + iz) * cd.IH + iy) * cd.IW + ix];
      }
      bT[tid][j] = v;
      if (++kx == cd.KW) {
        kx = 0;
        if (++ky == cd.KH) {
          ky = 0;
          if (++kd == cd.KD) { kd = 0; ++ci; }
        }
      }
    }
    __syncthreads();

    // ---------------- 4 WMMAs: 2 k-chunks x 2 N-subtiles -------------------
    {
      const int mrow = lane & 15;
      const int kbA  = (lane >> 4) * 16;   // A: lanes0-15 K=0..15, 16-31 K=16..31
      const int nn0  = (wave << 5) + (lane & 15);
      const int kbB  = (lane >> 4) * 16;   // B: lanes0-15 K=0..15, 16-31 K=16..31
      #pragma unroll
      for (int ks = 0; ks < 64; ks += 32) {
        v16h af, bf0, bf1;
        #pragma unroll
        for (int e = 0; e < 8; ++e) {
          af[e]     = aT[mrow][ks + kbA + e];
          af[e + 8] = aT[mrow][ks + kbA + 8 + e];
        }
        #pragma unroll
        for (int e = 0; e < 16; ++e) {
          bf0[e] = bT[nn0][ks + kbB + e];
          bf1[e] = bT[nn0 + 16][ks + kbB + e];
        }
        acc0 = __builtin_amdgcn_wmma_f32_16x16x32_f16(false, af, false, bf0,
                                                      (short)0, acc0, false, false);
        acc1 = __builtin_amdgcn_wmma_f32_16x16x32_f16(false, af, false, bf1,
                                                      (short)0, acc1, false, false);
      }
    }
    __syncthreads();
  }

  // --- write D: VGPR r -> M = r + 8*(lane>=16), N = lane%16 ---
  const int nA = tileN + (wave << 5) + (lane & 15);
  const int nB = nA + 16;
  const int mb = (lane >> 4) << 3;
  #pragma unroll
  for (int r = 0; r < 8; ++r) {
    int gm = m0 + mb + r;
    if (gm < cd.Co) {
      float b = bias[gm];
      if (nA < NSP) {
        float v = acc0[r] + b;
        if (cd.relu) v = fmaxf(v, 0.f);
        unsigned long long off = (unsigned long long)gm * NSP + nA;
        if (out16) out16[off] = (_Float16)v;
        if (out32) out32[off] = v;
      }
      if (nB < NSP) {
        float v = acc1[r] + b;
        if (cd.relu) v = fmaxf(v, 0.f);
        unsigned long long off = (unsigned long long)gm * NSP + nB;
        if (out16) out16[off] = (_Float16)v;
        if (out32) out32[off] = v;
      }
    }
  }
}

// ------------------------- transposed conv3d (gather) ----------------------
// deconv3d: lhs_dilation=2, pad (1,2), k=3  =>  out = 2*in, relu fused.

struct DeconvDims { int Ci, Co, ID, IH, IW, OD, OH, OW; };

__global__ void k_deconv3d(const _Float16* __restrict__ in,   // [Ci][ID][IH][IW]
                           const _Float16* __restrict__ wgt,  // [Co][Ci][27]
                           const float*    __restrict__ bias,
                           _Float16*       __restrict__ out,  // [Co][OD][OH][OW]
                           DeconvDims dd)
{
  size_t total = (size_t)dd.Co * dd.OD * dd.OH * dd.OW;
  size_t i = (size_t)blockIdx.x * blockDim.x + threadIdx.x;
  if (i >= total) return;
  int x = (int)(i % dd.OW); size_t r = i / dd.OW;
  int y = (int)(r % dd.OH); r /= dd.OH;
  int z = (int)(r % dd.OD);
  int co = (int)(r / dd.OD);
  float acc = bias[co];
  for (int kd = 0; kd < 3; ++kd) {
    int zd = z - 1 + kd;
    if (zd < 0 || (zd & 1) || (zd >> 1) >= dd.ID) continue;
    int iz = zd >> 1;
    for (int ky = 0; ky < 3; ++ky) {
      int yd = y - 1 + ky;
      if (yd < 0 || (yd & 1) || (yd >> 1) >= dd.IH) continue;
      int iy = yd >> 1;
      for (int kx = 0; kx < 3; ++kx) {
        int xd = x - 1 + kx;
        if (xd < 0 || (xd & 1) || (xd >> 1) >= dd.IW) continue;
        int ix = xd >> 1;
        int tap = kd * 9 + ky * 3 + kx;
        const _Float16* ip = in + ((size_t)iz * dd.IH + iy) * dd.IW + ix;
        const _Float16* wp = wgt + (size_t)co * dd.Ci * 27 + tap;
        size_t istride = (size_t)dd.ID * dd.IH * dd.IW;
        for (int ci = 0; ci < dd.Ci; ++ci)
          acc += (float)wp[(size_t)ci * 27] * (float)ip[(size_t)ci * istride];
      }
    }
  }
  out[i] = (_Float16)fmaxf(acc, 0.f);
}

// -------------------- plane-sweep fetch + variance cost --------------------
// feats: f32 [3][64][64][80]; cost: f16 [64][48][64][80]. View 0 = reference.

__global__ void k_cost(const float* __restrict__ feats, const float* __restrict__ pb,
                       _Float16* __restrict__ cost)
{
  const int C = 64, D = 48, FH = 64, FW = 80;
  const int HW = FH * FW;
  int idx = blockIdx.x * blockDim.x + threadIdx.x;
  if (idx >= D * HW * (C / 8)) return;
  int chunk = idx & 7;          // 8 channel-chunks of 8
  int pnt = idx >> 3;
  int x = pnt % FW;
  int t = pnt / FW;
  int y = t % FH;
  int d = t / FH;
  float ds = pb[36], di = pb[37];
  float depth = ds + (float)d * di;
  float px = x + 0.5f, py = y + 0.5f;

  int X0[2], X1[2], Y0[2], Y1[2];
  float WX[2], WY[2];
  #pragma unroll
  for (int v = 1; v <= 2; ++v) {
    const float* A = pb + v * 12;
    float Xc = depth * (A[0]*px + A[1]*py + A[2]) + A[9];
    float Yc = depth * (A[3]*px + A[4]*py + A[5]) + A[10];
    float Zc = depth * (A[6]*px + A[7]*py + A[8]) + A[11];
    float u = Xc / (Zc + 1e-8f);
    float w = Yc / (Zc + 1e-8f);
    u = fminf(fmaxf(u, 0.f), (float)(FW - 1));
    w = fminf(fmaxf(w, 0.f), (float)(FH - 1));
    float x0f = floorf(u), y0f = floorf(w);
    int x0 = (int)x0f, y0 = (int)y0f;
    X0[v-1] = x0;
    Y0[v-1] = y0;
    X1[v-1] = (x0 + 1 < FW) ? x0 + 1 : FW - 1;
    Y1[v-1] = (y0 + 1 < FH) ? y0 + 1 : FH - 1;
    WX[v-1] = u - x0f;
    WY[v-1] = w - y0f;
  }
  int cbase = chunk * 8;
  for (int cc = 0; cc < 8; ++cc) {
    int c = cbase + cc;
    float a = feats[(size_t)c * HW + y * FW + x];   // reference view
    float s = a, s2 = a * a;
    #pragma unroll
    for (int v = 0; v < 2; ++v) {
      const float* fv = feats + ((size_t)(v + 1) * C + c) * HW;
      float f00 = fv[Y0[v] * FW + X0[v]];
      float f01 = fv[Y0[v] * FW + X1[v]];
      float f10 = fv[Y1[v] * FW + X0[v]];
      float f11 = fv[Y1[v] * FW + X1[v]];
      float wx = WX[v], wy = WY[v];
      float val = f00 * (1.f-wx)*(1.f-wy) + f01 * wx*(1.f-wy)
                + f10 * (1.f-wx)*wy       + f11 * wx*wy;
      s += val; s2 += val * val;
    }
    float mean = s * (1.f / 3.f);
    float var  = s2 * (1.f / 3.f) - mean * mean;
    cost[(((size_t)c * D + d) * FH + y) * FW + x] = (_Float16)var;
  }
}

// ---------------- softmax over depth + regression + prob map ---------------

__global__ void k_depth(const float* __restrict__ filt, const float* __restrict__ pb,
                        float* __restrict__ outp)
{
  const int D = 48, HW = 64 * 80;
  int i = blockIdx.x * blockDim.x + threadIdx.x;
  if (i >= HW) return;
  float ds = pb[36], di = pb[37];
  float m = -3.4e38f;
  for (int d = 0; d < D; ++d) m = fmaxf(m, -filt[d * HW + i]);
  float se = 0.f, sed = 0.f;
  for (int d = 0; d < D; ++d) {
    float e = __expf(-filt[d * HW + i] - m);
    se  += e;
    sed += e * (ds + (float)d * di);
  }
  float depth = sed / se;
  outp[i] = depth;                               // depth_map
  int i0 = (int)floorf((depth - ds) / di);
  float tot = 0.f;
  for (int off = -1; off <= 2; ++off) {
    int ii = i0 + off;
    ii = ii < 0 ? 0 : (ii > D - 1 ? D - 1 : ii);
    tot += __expf(-filt[ii * HW + i] - m) / se;
  }
  outp[HW + i] = tot;                            // prob_map
}

// ------------------------------- host driver -------------------------------
// d_in order (setup_inputs dict order; params flattened in sorted-key order):
//  0 img_list  1 cam_params_list  2..45 params  46 isFlow  47 isTest

extern "C" void kernel_launch(void* const* d_in, const int* in_sizes, int n_in,
                              void* d_out, int out_size, void* d_ws, size_t ws_size,
                              hipStream_t stream) {
  (void)in_sizes; (void)n_in; (void)out_size; (void)ws_size;
  const float* img  = (const float*)d_in[0];
  const float* cams = (const float*)d_in[1];

  const int H0 = 512, W0 = 640;
  const int D3 = 48, H3 = 64, W3 = 80;

  char* ws = (char*)d_ws;
  size_t off = 0;
  auto alloc = [&](size_t bytes) -> void* {
    off = (off + 255) & ~(size_t)255;
    void* p = ws + off;
    off += bytes;
    return p;
  };

  float*     pb    = (float*)alloc(64 * sizeof(float));
  _Float16*  img16 = (_Float16*)alloc((size_t)3 * 3 * H0 * W0 * 2);
  _Float16*  bufA  = (_Float16*)alloc((size_t)8 * H0 * W0 * 2);
  _Float16*  bufB  = (_Float16*)alloc((size_t)8 * H0 * W0 * 2);
  float*     feats = (float*)alloc((size_t)3 * 64 * H3 * W3 * 4);
  _Float16*  cost  = (_Float16*)alloc((size_t)64 * D3 * H3 * W3 * 2);

  const size_t n0 = (size_t)D3 * H3 * W3;        // 245760 voxels (level 0)
  const size_t n1 = n0 / 8, n2 = n0 / 64, nb = n0 / 512;
  _Float16* c0_1 = (_Float16*)alloc(8  * n0 * 2);
  _Float16* c1_0 = (_Float16*)alloc(16 * n1 * 2);
  _Float16* c1_1 = (_Float16*)alloc(16 * n1 * 2);
  _Float16* c2_0 = (_Float16*)alloc(32 * n2 * 2);
  _Float16* c2_1 = (_Float16*)alloc(32 * n2 * 2);
  _Float16* c3_0 = (_Float16*)alloc(64 * nb * 2);
  _Float16* c3_1 = (_Float16*)alloc(64 * nb * 2);
  _Float16* c4_0 = (_Float16*)alloc(32 * n2 * 2);
  _Float16* sum2 = (_Float16*)alloc(32 * n2 * 2);
  _Float16* c5_0 = (_Float16*)alloc(16 * n1 * 2);
  _Float16* sum1 = (_Float16*)alloc(16 * n1 * 2);
  _Float16* c6_0 = (_Float16*)alloc(8  * n0 * 2);
  _Float16* sum0 = (_Float16*)alloc(8  * n0 * 2);
  float*    filt = (float*)alloc(n0 * 4);

  auto cvt = [&](const void* src, _Float16* dst, int n) {
    k_f32_to_f16<<<(n + 255) / 256, 256, 0, stream>>>((const float*)src, dst, n);
  };
  auto cvtw = [&](int idx, int n) -> const _Float16* {
    _Float16* p = (_Float16*)alloc((size_t)n * 2);
    cvt(d_in[idx], p, n);
    return p;
  };

  // geometry
  k_setup<<<1, 1, 0, stream>>>(cams, pb);

  // weights -> f16
  const _Float16* w_c0a = cvtw(3,  8*3*9);
  const _Float16* w_c0b = cvtw(5,  8*8*9);
  const _Float16* w_c1a = cvtw(7,  16*8*25);
  const _Float16* w_c1b = cvtw(9,  16*16*9);
  const _Float16* w_c1c = cvtw(11, 16*16*9);
  const _Float16* w_c2a = cvtw(13, 32*16*25);
  const _Float16* w_c2b = cvtw(15, 32*32*9);
  const _Float16* w_c2c = cvtw(17, 32*32*9);
  const _Float16* w_c3a = cvtw(19, 64*32*25);
  const _Float16* w_c3b = cvtw(21, 64*64*9);
  const _Float16* w_c3c = cvtw(23, 64*64*9);
  const _Float16* w_v01 = cvtw(25, 8*64*27);
  const _Float16* w_v10 = cvtw(27, 16*64*27);
  const _Float16* w_v11 = cvtw(29, 16*16*27);
  const _Float16* w_v20 = cvtw(31, 32*16*27);
  const _Float16* w_v21 = cvtw(33, 32*32*27);
  const _Float16* w_v30 = cvtw(35, 64*32*27);
  const _Float16* w_v31 = cvtw(37, 64*64*27);
  const _Float16* w_v40 = cvtw(39, 32*64*27);
  const _Float16* w_v50 = cvtw(41, 16*32*27);
  const _Float16* w_v60 = cvtw(43, 8*16*27);
  const _Float16* w_v62 = cvtw(45, 1*8*27);
  auto Bp = [&](int idx) { return (const float*)d_in[idx]; };

  cvt(img, img16, 3 * 3 * H0 * W0);

  auto conv = [&](const _Float16* in, const _Float16* w, const float* b,
                  _Float16* o16, float* o32,
                  int Ci, int Co, int KD, int KH, int KW,
                  int ID, int IH, int IW, int SD, int SH, int SW, int relu) {
    ConvDims cd;
    cd.Ci = Ci; cd.Co = Co;
    cd.KD = KD; cd.KH = KH; cd.KW = KW;
    cd.ID = ID; cd.IH = IH; cd.IW = IW;
    cd.PD = KD / 2; cd.PH = KH / 2; cd.PW = KW / 2;
    cd.SD = SD; cd.SH = SH; cd.SW = SW;
    cd.OD = (ID + 2 * cd.PD - KD) / SD + 1;
    cd.OH = (IH + 2 * cd.PH - KH) / SH + 1;
    cd.OW = (IW + 2 * cd.PW - KW) / SW + 1;
    cd.relu = relu;
    int NSP = cd.OD * cd.OH * cd.OW;
    dim3 grid((NSP + 255) / 256, (Co + 15) / 16);
    conv_wmma_kernel<<<grid, 256, 0, stream>>>(in, w, b, o16, o32, cd);
  };
  auto conv2 = [&](const _Float16* in, const _Float16* w, const float* b,
                   _Float16* o16, float* o32,
                   int Ci, int Co, int k, int IH, int IW, int s, int relu) {
    conv(in, w, b, o16, o32, Ci, Co, 1, k, k, 1, IH, IW, 1, s, s, relu);
  };
  auto deconv = [&](const _Float16* in, const _Float16* w, const float* b,
                    _Float16* o, int Ci, int Co, int ID, int IH, int IW) {
    DeconvDims dd;
    dd.Ci = Ci; dd.Co = Co;
    dd.ID = ID; dd.IH = IH; dd.IW = IW;
    dd.OD = 2 * ID; dd.OH = 2 * IH; dd.OW = 2 * IW;
    size_t total = (size_t)Co * dd.OD * dd.OH * dd.OW;
    k_deconv3d<<<(unsigned)((total + 255) / 256), 256, 0, stream>>>(in, w, b, o, dd);
  };
  auto add = [&](const _Float16* a, const _Float16* b, _Float16* o, size_t n) {
    k_add_f16<<<(unsigned)((n + 255) / 256), 256, 0, stream>>>(a, b, o, (int)n);
  };

  // -------- image CNN per view --------
  for (int v = 0; v < 3; ++v) {
    const _Float16* im = img16 + (size_t)v * 3 * H0 * W0;
    conv2(im,   w_c0a, Bp(2),  bufA, nullptr, 3,  8,  3, 512, 640, 1, 1);
    conv2(bufA, w_c0b, Bp(4),  bufB, nullptr, 8,  8,  3, 512, 640, 1, 1);
    conv2(bufB, w_c1a, Bp(6),  bufA, nullptr, 8,  16, 5, 512, 640, 2, 1);
    conv2(bufA, w_c1b, Bp(8),  bufB, nullptr, 16, 16, 3, 256, 320, 1, 1);
    conv2(bufB, w_c1c, Bp(10), bufA, nullptr, 16, 16, 3, 256, 320, 1, 1);
    conv2(bufA, w_c2a, Bp(12), bufB, nullptr, 16, 32, 5, 256, 320, 2, 1);
    conv2(bufB, w_c2b, Bp(14), bufA, nullptr, 32, 32, 3, 128, 160, 1, 1);
    conv2(bufA, w_c2c, Bp(16), bufB, nullptr, 32, 32, 3, 128, 160, 1, 1);
    conv2(bufB, w_c3a, Bp(18), bufA, nullptr, 32, 64, 5, 128, 160, 2, 1);
    conv2(bufA, w_c3b, Bp(20), bufB, nullptr, 64, 64, 3, 64,  80,  1, 1);
    conv2(bufB, w_c3c, Bp(22), nullptr, feats + (size_t)v * 64 * H3 * W3,
          64, 64, 3, 64, 80, 1, 0);                       // relu=False, f32 out
  }

  // -------- plane sweep + variance cost volume --------
  {
    int total = D3 * H3 * W3 * 8;
    k_cost<<<(total + 255) / 256, 256, 0, stream>>>(feats, pb, cost);
  }

  // -------- 3D U-Net --------
  conv(cost, w_v01, Bp(24), c0_1, nullptr, 64, 8,  3,3,3, 48,64,80, 1,1,1, 1);
  conv(cost, w_v10, Bp(26), c1_0, nullptr, 64, 16, 3,3,3, 48,64,80, 2,2,2, 1);
  conv(c1_0, w_v20, Bp(30), c2_0, nullptr, 16, 32, 3,3,3, 24,32,40, 2,2,2, 1);
  conv(c2_0, w_v30, Bp(34), c3_0, nullptr, 32, 64, 3,3,3, 12,16,20, 2,2,2, 1);
  conv(c1_0, w_v11, Bp(28), c1_1, nullptr, 16, 16, 3,3,3, 24,32,40, 1,1,1, 1);
  conv(c2_0, w_v21, Bp(32), c2_1, nullptr, 32, 32, 3,3,3, 12,16,20, 1,1,1, 1);
  conv(c3_0, w_v31, Bp(36), c3_1, nullptr, 64, 64, 3,3,3, 6, 8, 10, 1,1,1, 1);

  deconv(c3_1, w_v40, Bp(38), c4_0, 64, 32, 6, 8, 10);
  add(c4_0, c2_1, sum2, 32 * n2);
  deconv(sum2, w_v50, Bp(40), c5_0, 32, 16, 12, 16, 20);
  add(c5_0, c1_1, sum1, 16 * n1);
  deconv(sum1, w_v60, Bp(42), c6_0, 16, 8, 24, 32, 40);
  add(c6_0, c0_1, sum0, 8 * n0);

  conv(sum0, w_v62, Bp(44), nullptr, filt, 8, 1, 3,3,3, 48,64,80, 1,1,1, 0);

  // -------- softmax depth regression + prob map --------
  k_depth<<<(H3 * W3 + 255) / 256, 256, 0, stream>>>(filt, pb, (float*)d_out);
}
